// LSTMDecoder_15461882266217
// MI455X (gfx1250) — compile-verified, matrix operand order checked by reading, not run on hardware
//
#include <hip/hip_runtime.h>
#include <hip/hip_bf16.h>
#include <stdint.h>

// ---------------- problem constants ----------------
#define HD    1024     // hidden dim
#define BATCH 4096     // batch
#define TT    128      // forecast length
#define G4H   (4*HD)

// ---------------- tiling ----------------
#define BM 128         // batch rows per block (8 waves x 16)
#define BN 32          // hidden units per block -> 128 gate columns
#define BK 64          // K elements staged per LDS stage
#define NSTAGE (HD / BK)         // 16 stages
#define LROW 72                  // padded LDS row stride (elements) -> conflict-free
#define JROWS 128                // gate rows per stage tile (4 gates * 32 units)

typedef __attribute__((ext_vector_type(16))) __bf16 v16bf;
typedef __attribute__((ext_vector_type(8)))  __bf16 v8bf;
typedef __attribute__((ext_vector_type(8)))  float  v8f;
typedef int v4i __attribute__((vector_size(16)));

#if __has_builtin(__builtin_amdgcn_global_load_async_to_lds_b128) && \
    __has_builtin(__builtin_amdgcn_s_wait_asynccnt)
#define USE_ASYNC 1
#else
#define USE_ASYNC 0
#endif

__device__ __forceinline__ __bf16 f2bf(float f) {
    union { float f; uint32_t u; } v; v.f = f;
    uint32_t r = v.u + 0x7FFFu + ((v.u >> 16) & 1u);   // round-to-nearest-even
    union { uint16_t u; __bf16 b; } o; o.u = (uint16_t)(r >> 16);
    return o.b;
}

__device__ __forceinline__ float sigmoidf_(float x) {
    return 1.0f / (1.0f + __expf(-x));
}

__device__ __forceinline__ v16bf ld_frag(const __bf16* p) {   // global or LDS
    v8bf a0 = *(const v8bf*)(p);
    v8bf a1 = *(const v8bf*)(p + 16);
    return __builtin_shufflevector(a0, a1, 0,1,2,3,4,5,6,7,8,9,10,11,12,13,14,15);
}

#if USE_ASYNC
__device__ __forceinline__ void async_cp16(const __bf16* g, __bf16* l) {
    __builtin_amdgcn_global_load_async_to_lds_b128(
        (__attribute__((address_space(1))) v4i*)g,
        (__attribute__((address_space(3))) v4i*)l,
        0, 0);
}
#endif

// Stage one BK=64 chunk of the block's W_hh tile (128 gate rows x 64 K) into LDS.
// 256 threads, 2 threads per row, 32 elements (64 B = 4 x b128) per thread.
__device__ __forceinline__ void stage_B(const __bf16* __restrict__ Whh, int nbase, int k0,
                                        __bf16* __restrict__ smem, int tid) {
    const int jloc = tid >> 1;               // 0..127 : g*32 + (unit within 32)
    const int koff = (tid & 1) * 32;         // 0 or 32 elements
    const __bf16* g = Whh + ((size_t)(jloc >> 5) * HD + nbase + (jloc & 31)) * HD + k0 + koff;
    __bf16* l = smem + (size_t)jloc * LROW + koff;
#if USE_ASYNC
    async_cp16(g + 0,  l + 0);
    async_cp16(g + 8,  l + 8);
    async_cp16(g + 16, l + 16);
    async_cp16(g + 24, l + 24);
#else
    v8bf t0 = *(const v8bf*)(g + 0);
    v8bf t1 = *(const v8bf*)(g + 8);
    v8bf t2 = *(const v8bf*)(g + 16);
    v8bf t3 = *(const v8bf*)(g + 24);
    *(v8bf*)(l + 0)  = t0;
    *(v8bf*)(l + 8)  = t1;
    *(v8bf*)(l + 16) = t2;
    *(v8bf*)(l + 24) = t3;
#endif
}

// ---------------- prep kernels ----------------
__global__ void k_cvt_bf16(const float* __restrict__ in, __bf16* __restrict__ o, int n) {
    int i = blockIdx.x * blockDim.x + threadIdx.x;
    if (i < n) o[i] = f2bf(in[i]);
}

__global__ void k_bias(const float* __restrict__ bi, const float* __restrict__ bh,
                       float* __restrict__ bias, int n) {
    int i = blockIdx.x * blockDim.x + threadIdx.x;
    if (i < n) bias[i] = bi[i] + bh[i];
}

__global__ void k_init_out(float* __restrict__ out, const float* __restrict__ b_out, int t) {
    int b = blockIdx.x * blockDim.x + threadIdx.x;
    if (b < BATCH) out[(size_t)b * TT + t] = b_out[0];
}

// ---------------- fused LSTM step ----------------
// grid = (BATCH/BM, HD/BN) = (32, 32), block = 256 threads (8 waves).
// Each wave: 16 batch rows x 32 units x 4 gates -> 8 WMMA accumulators.
// W_hh tile double-buffered in LDS via async copies; A (h) streamed from L2.
__global__ __launch_bounds__(256)
void k_step(const __bf16* __restrict__ hin,    // [BATCH, HD] bf16 (prev h)
            const __bf16* __restrict__ Whh,    // [4H, HD] bf16 row-major
            const float*  __restrict__ bias,   // [4H]  (b_ih + b_hh)
            const float*  __restrict__ Wih,    // [4H]  (IN==1)
            const float*  __restrict__ Wout,   // [HD]
            const float*  __restrict__ cin,    // [BATCH, HD] f32 (prev c)
            float*        __restrict__ cout_,  // [BATCH, HD] f32 (new c)
            __bf16*       __restrict__ hout,   // [BATCH, HD] bf16 (new h)
            float*        __restrict__ out,    // [BATCH, TT] predictions (x feedback)
            int t)
{
    __shared__ __bf16 Bsm[2][JROWS * LROW];    // 2 x 18 KB

    const int tid   = threadIdx.x;
    const int lane  = tid & 31;
    const int wave  = tid >> 5;                     // 0..7
    const int mbase = blockIdx.x * BM + wave * 16;  // batch-row base of this wave
    const int nbase = blockIdx.y * BN;              // hidden-unit base of this block

    const int r16   = lane & 15;
    const int khalf = lane >> 4;

    const __bf16* arow = hin + (size_t)(mbase + r16) * HD + 8 * khalf;

    v8f acc[4][2] = {};

    stage_B(Whh, nbase, 0, Bsm[0], tid);

    for (int s = 0; s < NSTAGE; ++s) {
        const int cur = s & 1;
        if (s + 1 < NSTAGE) {
            stage_B(Whh, nbase, (s + 1) * BK, Bsm[cur ^ 1], tid);
#if USE_ASYNC
            __builtin_amdgcn_s_wait_asynccnt(4);   // drain current stage, keep next in flight
#endif
        } else {
#if USE_ASYNC
            __builtin_amdgcn_s_wait_asynccnt(0);   // last stage: drain everything
#endif
        }
        __syncthreads();

        const int k0 = s * BK;
        __builtin_prefetch(arow + k0 + BK, 0, 1);   // next A stage

        #pragma unroll
        for (int ks = 0; ks < 2; ++ks) {
            v16bf a = ld_frag(arow + k0 + ks * 32);
            #pragma unroll
            for (int g = 0; g < 4; ++g) {
                #pragma unroll
                for (int u = 0; u < 2; ++u) {
                    const __bf16* bp = &Bsm[cur][(size_t)(g * 32 + u * 16 + r16) * LROW
                                                 + ks * 32 + 8 * khalf];
                    v16bf b = ld_frag(bp);
                    acc[g][u] = __builtin_amdgcn_wmma_f32_16x16x32_bf16(
                        false, a, false, b, (short)0, acc[g][u], false, false);
                }
            }
        }
        __syncthreads();
    }

    // ---- fused elementwise epilogue ----
    // D layout: VGPR v, lanes 0-15 -> M=v, N=lane; lanes 16-31 -> M=v+8, N=lane-16.
    float wi[4][2], bs[4][2], wo[2];
    #pragma unroll
    for (int u = 0; u < 2; ++u) {
        const int n = nbase + u * 16 + r16;
        wo[u] = Wout[n];
        #pragma unroll
        for (int g = 0; g < 4; ++g) {
            wi[g][u] = Wih[g * HD + n];
            bs[g][u] = bias[g * HD + n];
        }
    }

    #pragma unroll
    for (int v = 0; v < 8; ++v) {
        const int row = mbase + v + 8 * khalf;
        const float x = (t == 0) ? 0.0f : out[(size_t)row * TT + (t - 1)];

        #pragma unroll
        for (int u = 0; u < 2; ++u) {
            const int n = nbase + u * 16 + r16;

            const float gi = acc[0][u][v] + bs[0][u] + x * wi[0][u];
            const float gf = acc[1][u][v] + bs[1][u] + x * wi[1][u];
            const float gg = acc[2][u][v] + bs[2][u] + x * wi[2][u];
            const float go = acc[3][u][v] + bs[3][u] + x * wi[3][u];

            const float cold = cin[(size_t)row * HD + n];
            const float cnew = sigmoidf_(gf) * cold + sigmoidf_(gi) * tanhf(gg);
            const float hnew = sigmoidf_(go) * tanhf(cnew);

            cout_[(size_t)row * HD + n] = cnew;
            hout [(size_t)row * HD + n] = f2bf(hnew);

            // pred partial: reduce h*wout over the 16 columns of this unit sub-tile
            float p = hnew * wo[u];
            p += __shfl_xor(p, 8, 16);
            p += __shfl_xor(p, 4, 16);
            p += __shfl_xor(p, 2, 16);
            p += __shfl_xor(p, 1, 16);
            if (r16 == 0) atomicAdd(&out[(size_t)row * TT + t], p);
        }
    }
}

// ---------------- host orchestration ----------------
extern "C" void kernel_launch(void* const* d_in, const int* in_sizes, int n_in,
                              void* d_out, int out_size, void* d_ws, size_t ws_size,
                              hipStream_t stream) {
    (void)in_sizes; (void)n_in; (void)out_size; (void)ws_size;

    const float* hidden = (const float*)d_in[0];   // [1,B,H]
    const float* cell   = (const float*)d_in[1];   // [1,B,H]
    const float* W_ih   = (const float*)d_in[2];   // [4H,1] -> [4H]
    const float* W_hh   = (const float*)d_in[3];   // [4H,H]
    const float* b_ih   = (const float*)d_in[4];   // [4H]
    const float* b_hh   = (const float*)d_in[5];   // [4H]
    const float* W_out  = (const float*)d_in[6];   // [1,H] -> [H]
    const float* b_out  = (const float*)d_in[7];   // [1]
    float* out = (float*)d_out;                    // [B,T,1]

    // workspace layout
    char* ws = (char*)d_ws;
    __bf16* Whh_bf = (__bf16*)ws;                                   //  8 MB
    __bf16* hbuf0  = (__bf16*)(ws + (size_t)8  * 1024 * 1024);      //  8 MB
    __bf16* hbuf1  = hbuf0 + (size_t)BATCH * HD;                    //  8 MB
    float*  c_ws   = (float*) (ws + (size_t)24 * 1024 * 1024);      // 16 MB
    float*  bias   = (float*) (ws + (size_t)40 * 1024 * 1024);      // 16 KB

    const int nW = G4H * HD;
    const int nS = BATCH * HD;

    k_cvt_bf16<<<(nW + 255) / 256, 256, 0, stream>>>(W_hh, Whh_bf, nW);
    k_cvt_bf16<<<(nS + 255) / 256, 256, 0, stream>>>(hidden, hbuf0, nS);
    k_bias<<<(G4H + 255) / 256, 256, 0, stream>>>(b_ih, b_hh, bias, G4H);

    dim3 grid(BATCH / BM, HD / BN);   // 32 x 32
    for (int t = 0; t < TT; ++t) {
        k_init_out<<<BATCH / 256, 256, 0, stream>>>(out, b_out, t);
        const __bf16* hin  = (t & 1) ? hbuf1 : hbuf0;
        __bf16*       hout = (t & 1) ? hbuf0 : hbuf1;
        const float*  cin  = (t == 0) ? cell : c_ws;
        k_step<<<grid, 256, 0, stream>>>(hin, Whh_bf, bias, W_ih, W_out,
                                         cin, c_ws, hout, out, t);
    }
}
